// BertSelfOutput_61813169324800
// MI455X (gfx1250) — 8-bit matrix operand order repaired and checked by reading, compiled, not run on hardware
//
#include <hip/hip_runtime.h>

typedef int v8i __attribute__((ext_vector_type(8)));

#define HDIM  1024
#define CLIPV 2.5f
#define QMAX  127.0f
#define LNEPS 1e-12f
#define BM    64
#define WN    64
#define KSTEP 64

__global__ void init_slots_kernel(unsigned* slots) {
  if (threadIdx.x < 2) slots[threadIdx.x] = 0u;
}

__global__ void absmax_kernel(const float* __restrict__ x, unsigned n,
                              unsigned* __restrict__ slot) {
  float m = 0.0f;
  const unsigned stride = gridDim.x * blockDim.x;
  for (unsigned i = blockIdx.x * blockDim.x + threadIdx.x; i < n; i += stride) {
    float v = fminf(fmaxf(x[i], -CLIPV), CLIPV);
    m = fmaxf(m, fabsf(v));
  }
#pragma unroll
  for (int off = 16; off > 0; off >>= 1)
    m = fmaxf(m, __shfl_xor(m, off, 32));
  if ((threadIdx.x & 31) == 0) atomicMax(slot, __float_as_uint(m));
}

__global__ void quant4_kernel(const float4* __restrict__ x, unsigned n4,
                              const unsigned* __restrict__ slot,
                              char4* __restrict__ q) {
  const float m = __uint_as_float(*slot);
  const float s = QMAX / fmaxf(m, 1e-12f);
  const unsigned stride = gridDim.x * blockDim.x;
  for (unsigned i = blockIdx.x * blockDim.x + threadIdx.x; i < n4; i += stride) {
    float4 v = x[i];
    char4 o;
    o.x = (signed char)(int)rintf(fminf(fmaxf(v.x, -CLIPV), CLIPV) * s);
    o.y = (signed char)(int)rintf(fminf(fmaxf(v.y, -CLIPV), CLIPV) * s);
    o.z = (signed char)(int)rintf(fminf(fmaxf(v.z, -CLIPV), CLIPV) * s);
    o.w = (signed char)(int)rintf(fminf(fmaxf(v.w, -CLIPV), CLIPV) * s);
    q[i] = o;
  }
}

__global__ __launch_bounds__(512, 1)
void gemm_ln_kernel(const signed char* __restrict__ xq,
                    const signed char* __restrict__ wq,
                    const unsigned*    __restrict__ slots,
                    const float*       __restrict__ bias,
                    const float*       __restrict__ resid,
                    const float*       __restrict__ gamma,
                    const float*       __restrict__ beta,
                    float*             __restrict__ out) {
  __shared__ __align__(16) signed char sA[BM * KSTEP];
  __shared__ float s_sum[BM];
  __shared__ float s_sq[BM];
  __shared__ float s_mu[BM];
  __shared__ float s_rs[BM];

  const int tid  = threadIdx.x;
  const int wave = tid >> 5;
  const int lane = tid & 31;
  const int half = lane >> 4;
  const int l16  = lane & 15;
  const int m_blk  = blockIdx.x * BM;
  const int n_wave = wave * WN;

  v8i acc[4][4];
#pragma unroll
  for (int mi = 0; mi < 4; ++mi)
#pragma unroll
    for (int ni = 0; ni < 4; ++ni) acc[mi][ni] = (v8i)0;

  const int sr = tid >> 3;
  const int sc = (tid & 7) * 8;
  const size_t xrow = (size_t)(m_blk + sr) * HDIM + sc;

  for (int kb = 0; kb < HDIM; kb += KSTEP) {
    __syncthreads();
    *(unsigned long long*)&sA[sr * KSTEP + sc] =
        *(const unsigned long long*)&xq[xrow + kb];
    if (kb + KSTEP < HDIM)
      __builtin_prefetch(&wq[(size_t)(n_wave + l16) * HDIM + kb + KSTEP], 0, 3);
    __syncthreads();

    v8i a[4];
#pragma unroll
    for (int mi = 0; mi < 4; ++mi) {
      const signed char* p = &sA[(mi * 16 + l16) * KSTEP + half * 16];
      int2 q0 = *(const int2*)(p + 0);
      int2 q1 = *(const int2*)(p + 8);
      int2 q2 = *(const int2*)(p + 32);
      int2 q3 = *(const int2*)(p + 40);
      a[mi][0] = q0.x; a[mi][1] = q0.y; a[mi][2] = q1.x; a[mi][3] = q1.y;
      a[mi][4] = q2.x; a[mi][5] = q2.y; a[mi][6] = q3.x; a[mi][7] = q3.y;
    }

    v8i b[4];
#pragma unroll
    for (int ni = 0; ni < 4; ++ni) {
      const signed char* p =
          &wq[(size_t)(n_wave + ni * 16 + l16) * HDIM + kb + half * 16];
      int4 r0 = *(const int4*)(p + 0);
      int4 r1 = *(const int4*)(p + 32);
      b[ni][0] = r0.x; b[ni][1] = r0.y; b[ni][2] = r0.z; b[ni][3] = r0.w;
      b[ni][4] = r1.x; b[ni][5] = r1.y; b[ni][6] = r1.z; b[ni][7] = r1.w;
    }

#pragma unroll
    for (int mi = 0; mi < 4; ++mi)
#pragma unroll
      for (int ni = 0; ni < 4; ++ni)
        acc[mi][ni] = __builtin_amdgcn_wmma_i32_16x16x64_iu8(
            true, a[mi], true, b[ni], acc[mi][ni], false, false);
  }

  const float inv = (__uint_as_float(slots[0]) / QMAX) *
                    (__uint_as_float(slots[1]) / QMAX);

  float g[4], bt[4], bs[4];
#pragma unroll
  for (int ni = 0; ni < 4; ++ni) {
    const int nn = n_wave + ni * 16 + l16;
    bs[ni] = bias[nn];
    g[ni]  = gamma[nn];
    bt[ni] = beta[nn];
  }

  float yv[4][4][8];
#pragma unroll
  for (int mi = 0; mi < 4; ++mi)
#pragma unroll
    for (int r = 0; r < 8; ++r) {
      const int mm = mi * 16 + r + half * 8;
      const size_t rbase = (size_t)(m_blk + mm) * HDIM;
#pragma unroll
      for (int ni = 0; ni < 4; ++ni) {
        const int nn = n_wave + ni * 16 + l16;
        yv[mi][ni][r] =
            (float)acc[mi][ni][r] * inv + bs[ni] + resid[rbase + nn];
      }
    }

  if (tid < BM) { s_sum[tid] = 0.0f; s_sq[tid] = 0.0f; }
  __syncthreads();

#pragma unroll
  for (int mi = 0; mi < 4; ++mi)
#pragma unroll
    for (int r = 0; r < 8; ++r) {
      float s = 0.0f, q = 0.0f;
#pragma unroll
      for (int ni = 0; ni < 4; ++ni) {
        const float v = yv[mi][ni][r];
        s += v;
        q += v * v;
      }
#pragma unroll
      for (int off = 1; off < 16; off <<= 1) {
        s += __shfl_xor(s, off, 32);
        q += __shfl_xor(q, off, 32);
      }
      if (l16 == 0) {
        const int mm = mi * 16 + r + half * 8;
        atomicAdd(&s_sum[mm], s);
        atomicAdd(&s_sq[mm], q);
      }
    }
  __syncthreads();

  if (tid < BM) {
    const float mu  = s_sum[tid] * (1.0f / HDIM);
    const float var = s_sq[tid] * (1.0f / HDIM) - mu * mu;
    s_mu[tid] = mu;
    s_rs[tid] = rsqrtf(var + LNEPS);
  }
  __syncthreads();

#pragma unroll
  for (int mi = 0; mi < 4; ++mi)
#pragma unroll
    for (int r = 0; r < 8; ++r) {
      const int mm = mi * 16 + r + half * 8;
      const float mu = s_mu[mm];
      const float rs = s_rs[mm];
      const size_t rbase = (size_t)(m_blk + mm) * HDIM;
#pragma unroll
      for (int ni = 0; ni < 4; ++ni) {
        const int nn = n_wave + ni * 16 + l16;
        out[rbase + nn] = (yv[mi][ni][r] - mu) * rs * g[ni] + bt[ni];
      }
    }
}

extern "C" void kernel_launch(void* const* d_in, const int* in_sizes, int n_in,
                              void* d_out, int out_size, void* d_ws,
                              size_t ws_size, hipStream_t stream) {
  const float* hidden = (const float*)d_in[0];
  const float* resid  = (const float*)d_in[1];
  const float* weight = (const float*)d_in[2];
  const float* bias   = (const float*)d_in[3];
  const float* gamma  = (const float*)d_in[4];
  const float* beta   = (const float*)d_in[5];
  float* out = (float*)d_out;

  const unsigned NX = (unsigned)in_sizes[0];
  const unsigned NW = (unsigned)in_sizes[2];
  const unsigned M  = NX / HDIM;

  unsigned* slots = (unsigned*)d_ws;
  signed char* wq = (signed char*)d_ws + 256;
  signed char* xq = wq + (size_t)NW;

  init_slots_kernel<<<1, 32, 0, stream>>>(slots);
  absmax_kernel<<<1024, 256, 0, stream>>>(hidden, NX, slots + 0);
  absmax_kernel<<<256, 256, 0, stream>>>(weight, NW, slots + 1);
  quant4_kernel<<<2048, 256, 0, stream>>>((const float4*)hidden, NX / 4,
                                          slots + 0, (char4*)xq);
  quant4_kernel<<<512, 256, 0, stream>>>((const float4*)weight, NW / 4,
                                         slots + 1, (char4*)wq);
  gemm_ln_kernel<<<M / BM, 512, 0, stream>>>(xq, wq, slots, bias, resid,
                                             gamma, beta, out);
}
